// DP_DCNv2_48876727828660
// MI455X (gfx1250) — matrix operand order repaired and checked by reading, not compiled, not run on hardware
//
#include <hip/hip_runtime.h>
#include <hip/hip_bf16.h>

typedef _Float16 half_t;
typedef __attribute__((ext_vector_type(2)))  _Float16 v2h;
typedef __attribute__((ext_vector_type(16))) _Float16 v16h;
typedef __attribute__((ext_vector_type(8)))  float    v8f;

#define Bn 4
#define Gn 4
#define Hn 128
#define Wn 128
#define HW (Hn*Wn)
#define CK 576          // contraction: 64 channels * 9 taps (ck' = k*64 + c)
#define CHUNKS 18       // CK / 32
#define ROWH 592        // halves per pixel row in LDS (576 + 16 pad), 1184B (16B aligned)
#define TPX 32          // pixels per wave tile (two 16-wide WMMA N-tiles)

// ---------------------------------------------------------------------------
// NCHW f32 -> per-(b,g) channels-last f16: xt[(b*4+g)*HW + y*W + x][c]
// ---------------------------------------------------------------------------
__global__ void transpose_x(const float* __restrict__ x, half_t* __restrict__ xt) {
    __shared__ float tile[64][33];
    int bid = blockIdx.x;            // 8192 = b(4) * g(4) * y(128) * x32(4)
    int x32 = bid & 3;
    int y   = (bid >> 2) & 127;
    int g   = (bid >> 9) & 3;
    int b   = bid >> 11;
    int tid = threadIdx.x;
    #pragma unroll
    for (int i = 0; i < 8; ++i) {
        int e = tid + i * 256;
        int c = e >> 5, xx = e & 31;
        tile[c][xx] = x[((size_t)(b*256 + g*64 + c))*HW + y*Wn + x32*32 + xx];
    }
    __syncthreads();
    #pragma unroll
    for (int i = 0; i < 8; ++i) {
        int e = tid + i * 256;
        int c = e & 63, px = e >> 6;
        xt[((size_t)((b*Gn + g))*HW + y*Wn + x32*32 + px)*64 + c] = (half_t)tile[c][px];
    }
}

// ---------------------------------------------------------------------------
// Pre-swizzle weights into the WMMA 16-bit A-matrix (16x32) lane layout.
// Flat half index = ((((g*NOCT+oct)*18+chunk)*32)+lane)*16 + h
// lane: M = lane&15, hi = lane>>4 ; half h -> K = h<8 ? hi*8+h : 16+hi*8+(h-8)
// ck' = chunk*32 + K ; c = ck'&63 ; k = ck'>>6
// ---------------------------------------------------------------------------
__global__ void pack_w(const float* __restrict__ w, half_t* __restrict__ pw) {
    int idx = blockIdx.x * blockDim.x + threadIdx.x;   // 4*4*18*32*16 = 147456
    int h    = idx & 15;
    int lane = (idx >> 4) & 31;
    int r    = idx >> 9;
    int chunk = r % 18; r /= 18;
    int oct = r & 3;  int g = r >> 2;
    int m = lane & 15, hi = lane >> 4;
    int kk  = (h < 8) ? (hi*8 + h) : (16 + hi*8 + (h - 8));
    int ckp = chunk*32 + kk;
    int c = ckp & 63, k = ckp >> 6;
    int oc = oct*16 + m;
    pw[idx] = (half_t) w[((size_t)((g*64 + oc)*64 + c))*9 + k];
}

__global__ void pack_wom(const float* __restrict__ w, half_t* __restrict__ pw) {
    int idx = blockIdx.x * blockDim.x + threadIdx.x;   // 4*2*18*32*16 = 73728
    int h    = idx & 15;
    int lane = (idx >> 4) & 31;
    int r    = idx >> 9;
    int chunk = r % 18; r /= 18;
    int oct = r & 1;  int g = r >> 1;
    int m = lane & 15, hi = lane >> 4;
    int kk  = (h < 8) ? (hi*8 + h) : (16 + hi*8 + (h - 8));
    int ckp = chunk*32 + kk;
    int c = ckp & 63, k = ckp >> 6;
    int oc = oct*16 + m;
    float v = (oc < 27) ? w[((size_t)((g*27 + oc)*64 + c))*9 + k] : 0.0f;
    pw[idx] = (half_t)v;
}

// Fold bias + BatchNorm into per-channel v' = v*bnA + bnB
__global__ void bn_prep(const float* __restrict__ bias, const float* __restrict__ gamma,
                        const float* __restrict__ beta, const float* __restrict__ mean,
                        const float* __restrict__ var, float* __restrict__ bnA,
                        float* __restrict__ bnB) {
    int occ = blockIdx.x * blockDim.x + threadIdx.x;   // 256
    float sc = gamma[occ] * rsqrtf(var[occ] + 1e-5f);
    bnA[occ] = sc;
    bnB[occ] = (bias[occ] - mean[occ]) * sc + beta[occ];
}

// ---------------------------------------------------------------------------
// MODE 0: plain 3x3 conv 64->27 (offset/mask head), sigmoid on mask channels.
// MODE 1: modulated deformable conv 64->64, fused bias+BN (bnA/bnB) + SiLU.
// One wave per 32-pixel tile: deformed im2col staged in a private LDS slice
// s[32 px][9 k][64 c] (f16); each prepacked A fragment feeds TWO
// v_wmma_f32_16x16x32_f16 (two N-tiles), halving A-side VMEM per WMMA.
// ---------------------------------------------------------------------------
template<int MODE>
__global__ __launch_bounds__(32)
void dconv(const half_t* __restrict__ xt, const float* __restrict__ om,
           const v16h* __restrict__ packA, const float* __restrict__ bias,
           const float* __restrict__ bnA, const float* __restrict__ bnB,
           float* __restrict__ outp) {
    __shared__ __align__(16) half_t s[TPX * ROWH];     // 37,888 B
    __shared__ float4 gw[9 * TPX];                     //  4,608 B (w00..w11, premult by mask)
    __shared__ int2   gi[9 * TPX];                     //  2,304 B (iy, ix)
    int lane = threadIdx.x;
    int t = blockIdx.x;                  // 8192 tiles = g(4)*b(4)*y(128)*x4(4)
    int xt4 = t & 3;
    int y   = (t >> 2) & 127;
    int b   = (t >> 9) & 3;
    int g   = (t >> 11) & 3;
    int x0  = xt4 * TPX;
    const half_t* xb = xt + ((size_t)(b*Gn + g))*HW*64;
    int c0 = lane * 2;                   // each lane owns 2 channels

    // ---------------- Phase 1: fill LDS im2col tile ----------------
    if (MODE == 1) {
        const float* ob = om + ((size_t)(g*Bn + b))*27*HW;
        // 1a: geometry, lane-parallel over pixels (coalesced om reads)
        #pragma unroll
        for (int k = 0; k < 9; ++k) {
            int p = lane;                                // pixel within tile
            int pix = y*Wn + x0 + p;
            float dy = ob[k*HW + pix];
            float dx = ob[(9 + k)*HW + pix];
            float mm = ob[(18 + k)*HW + pix];            // already sigmoided
            float py  = (float)y        - 1.0f + (float)(k / 3) + dy;
            float pxf = (float)(x0 + p) - 1.0f + (float)(k % 3) + dx;
            float fy = floorf(py), fx = floorf(pxf);
            float wy1 = py - fy, wx1 = pxf - fx;
            float wy0 = 1.0f - wy1, wx0 = 1.0f - wx1;
            float4 wv; wv.x = wy0*wx0*mm; wv.y = wy0*wx1*mm;
                       wv.z = wy1*wx0*mm; wv.w = wy1*wx1*mm;
            gw[k*TPX + p] = wv;
            int2 ii; ii.x = (int)fy; ii.y = (int)fx;
            gi[k*TPX + p] = ii;
        }
        // 1b: channel gather, geometry broadcast from LDS (same wave => ordered)
        for (int p = 0; p < TPX; ++p) {
            #pragma unroll
            for (int k = 0; k < 9; ++k) {
                float4 wv = gw[k*TPX + p];
                int2   ii = gi[k*TPX + p];
                float a0 = 0.0f, a1 = 0.0f;
                #pragma unroll
                for (int cr = 0; cr < 4; ++cr) {
                    int yy = ii.x + (cr >> 1), xc = ii.y + (cr & 1);
                    float wgt = (cr == 0) ? wv.x : (cr == 1) ? wv.y : (cr == 2) ? wv.z : wv.w;
                    if (yy >= 0 && yy < Hn && xc >= 0 && xc < Wn) {
                        v2h pv = *(const v2h*)(xb + ((size_t)(yy*Wn + xc))*64 + c0);
                        a0 += wgt * (float)pv.x;
                        a1 += wgt * (float)pv.y;
                    }
                }
                v2h st; st.x = (half_t)a0; st.y = (half_t)a1;
                *(v2h*)(s + p*ROWH + k*64 + c0) = st;    // packed ds_store_b32
            }
        }
    } else {
        for (int p = 0; p < TPX; ++p) {
            int xx = x0 + p;
            #pragma unroll
            for (int k = 0; k < 9; ++k) {
                int yy = y - 1 + k / 3, xc = xx - 1 + k % 3;
                v2h pv; pv.x = (half_t)0.0f; pv.y = (half_t)0.0f;
                if (yy >= 0 && yy < Hn && xc >= 0 && xc < Wn)
                    pv = *(const v2h*)(xb + ((size_t)(yy*Wn + xc))*64 + c0);
                *(v2h*)(s + p*ROWH + k*64 + c0) = pv;
            }
        }
    }
    // Single-wave block: in-wave DScnt waits order fill -> consume; no barrier.

    // ---------------- Phase 2: WMMA GEMM over ck' = 576 ----------------
    const int noct = (MODE == 1) ? 4 : 2;
    int n = lane & 15, hi = lane >> 4;
    for (int oct = 0; oct < noct; ++oct) {
        v8f acc0 = {}, acc1 = {};
        const v16h* Ao = packA + ((size_t)(g*noct + oct))*CHUNKS*32 + lane;
        #pragma unroll
        for (int ch = 0; ch < CHUNKS; ++ch) {
            v16h a = Ao[ch*32];                            // prepacked A fragment
            int kb = ch*32 + hi*8;                         // B: lane n, K = kb..kb+7, kb+16..kb+23
            typedef __attribute__((ext_vector_type(8))) _Float16 v8h;
            v8h b0l = *(const v8h*)(s + n*ROWH + kb), b0h = *(const v8h*)(s + n*ROWH + kb + 16);
            v8h b1l = *(const v8h*)(s + (n + 16)*ROWH + kb), b1h = *(const v8h*)(s + (n + 16)*ROWH + kb + 16);
            v16h b0 = __builtin_shufflevector(b0l, b0h, 0,1,2,3,4,5,6,7,8,9,10,11,12,13,14,15);          // N-tile 0
            v16h b1 = __builtin_shufflevector(b1l, b1h, 0,1,2,3,4,5,6,7,8,9,10,11,12,13,14,15);   // N-tile 1
            acc0 = __builtin_amdgcn_wmma_f32_16x16x32_f16(
                       false, a, false, b0, (short)0, acc0, false, false);
            acc1 = __builtin_amdgcn_wmma_f32_16x16x32_f16(
                       false, a, false, b1, (short)0, acc1, false, false);
        }
        int px0 = x0 + n, px1 = x0 + 16 + n;
        if (MODE == 1) {
            #pragma unroll
            for (int r = 0; r < 8; ++r) {
                int occ = g*64 + oct*16 + hi*8 + r;        // C/D layout: M = r + 8*hi
                float A = bnA[occ], Bc = bnB[occ];
                size_t rowoff = ((size_t)(b*256 + occ))*HW + y*Wn;
                float v0 = acc0[r] * A + Bc;
                v0 = v0 * (1.0f / (1.0f + __expf(-v0)));   // SiLU
                outp[rowoff + px0] = v0;
                float v1 = acc1[r] * A + Bc;
                v1 = v1 * (1.0f / (1.0f + __expf(-v1)));
                outp[rowoff + px1] = v1;
            }
        } else {
            #pragma unroll
            for (int r = 0; r < 8; ++r) {
                int oc = oct*16 + hi*8 + r;
                if (oc < 27) {
                    float bb = bias[g*27 + oc];
                    size_t rowoff = ((size_t)(g*Bn + b)*27 + oc)*HW + y*Wn;
                    float v0 = acc0[r] + bb;
                    float v1 = acc1[r] + bb;
                    if (oc >= 18) {                         // sigmoid(mask logits)
                        v0 = 1.0f / (1.0f + __expf(-v0));
                        v1 = 1.0f / (1.0f + __expf(-v1));
                    }
                    outp[rowoff + px0] = v0;
                    outp[rowoff + px1] = v1;
                }
            }
        }
    }
}

// ---------------------------------------------------------------------------
extern "C" void kernel_launch(void* const* d_in, const int* in_sizes, int n_in,
                              void* d_out, int out_size, void* d_ws, size_t ws_size,
                              hipStream_t stream) {
    const float* x     = (const float*)d_in[0];
    const float* w     = (const float*)d_in[1];
    const float* bconv = (const float*)d_in[2];
    const float* w_om  = (const float*)d_in[3];
    const float* b_om  = (const float*)d_in[4];
    const float* gamma = (const float*)d_in[5];
    const float* beta  = (const float*)d_in[6];
    const float* mean  = (const float*)d_in[7];
    const float* var   = (const float*)d_in[8];
    float* out = (float*)d_out;

    char* ws = (char*)d_ws;
    half_t* xtb  = (half_t*)(ws);                                   // 33,554,432 B
    float*  omb  = (float*)(ws + 33554432);                         // 28,311,552 B
    half_t* pw   = (half_t*)(ws + 33554432 + 28311552);             //    294,912 B
    half_t* pwom = (half_t*)(ws + 33554432 + 28311552 + 294912);    //    147,456 B
    float*  bnA  = (float*)(ws + 33554432 + 28311552 + 294912 + 147456);  // 1024 B
    float*  bnB  = bnA + 256;                                             // 1024 B

    transpose_x<<<8192, 256, 0, stream>>>(x, xtb);
    pack_w    <<<576,  256, 0, stream>>>(w, pw);
    pack_wom  <<<288,  256, 0, stream>>>(w_om, pwom);
    bn_prep   <<<1,    256, 0, stream>>>(bconv, gamma, beta, mean, var, bnA, bnB);
    // Offset/mask head (plain conv via WMMA)
    dconv<0><<<8192, 32, 0, stream>>>(xtb, nullptr, (const v16h*)pwom,
                                      b_om, nullptr, nullptr, omb);
    // Deformable conv + fused BN + SiLU
    dconv<1><<<8192, 32, 0, stream>>>(xtb, omb, (const v16h*)pw,
                                      nullptr, bnA, bnB, out);
}
